// SCNN_65979287601855
// MI455X (gfx1250) — matrix operand order repaired and checked by reading, not compiled, not run on hardware
//
#include <hip/hip_runtime.h>

// ---------------- types ----------------
typedef unsigned short u16;
typedef __attribute__((ext_vector_type(16))) __bf16 v16bf;
typedef __attribute__((ext_vector_type(8)))  float  v8f;
typedef __attribute__((ext_vector_type(8)))  unsigned short v8u;
typedef __attribute__((ext_vector_type(16))) unsigned short v16u;

__device__ __forceinline__ u16 f32_to_bf16u(float f) {
  union { float f; unsigned int u; } v; v.f = f;
  unsigned int r = v.u + 0x7FFFu + ((v.u >> 16) & 1u);
  return (u16)(r >> 16);
}
__device__ __forceinline__ float bf16u_to_f32(u16 s) {
  union { unsigned int u; float f; } v; v.u = ((unsigned int)s) << 16;
  return v.f;
}

// ---------------- fp32 -> bf16 convert (same layout) ----------------
__global__ void cvt_f32_bf16(const float* __restrict__ in, u16* __restrict__ out, int n) {
  int i = blockIdx.x * blockDim.x + threadIdx.x;
  if (i < n) out[i] = f32_to_bf16u(in[i]);
}

// ---------------- NCHW fp32 -> NHWC bf16 (input image) ----------------
__global__ void cvt_nchw_nhwc(const float* __restrict__ in, u16* __restrict__ out,
                              int N, int C, int H, int W) {
  int idx = blockIdx.x * blockDim.x + threadIdx.x;
  int total = N * H * W * C;
  if (idx >= total) return;
  int c = idx % C; int t = idx / C;
  int w = t % W; t /= W;
  int h = t % H; int n = t / H;
  out[idx] = f32_to_bf16u(in[(((size_t)n * C + c) * H + h) * W + w]);
}

// ---------------- weight repack into WMMA B-fragment order ----------------
// GEMM k ordering = (kh, kw, cin), matching NHWC activations.
// B (32x16) bf16 fragment order used here (contiguous k, same as A): lane<16 -> N=lane, K=j; lane>=16 -> N=lane-16, K=16+j.
__global__ void repack_w(const float* __restrict__ w, u16* __restrict__ out,
                         int Cout, int Cin, int KH, int KW, int KB, int NB) {
  int total = KB * NB * 512;
  int idx = blockIdx.x * blockDim.x + threadIdx.x;
  if (idx >= total) return;
  int j    = idx & 15;
  int lane = (idx >> 4) & 31;
  int blk  = idx >> 9;
  int nb = blk % NB;
  int kb = blk / NB;
  int koff = (lane < 16) ? j : (16 + j);
  int k = kb * 32 + koff;
  int n = nb * 16 + (lane & 15);
  int K = Cin * KH * KW;
  float v = 0.f;
  if (k < K && n < Cout) {
    int khkw = k / Cin; int cin = k - khkw * Cin;
    int kh = khkw / KW; int kw = khkw - kh * KW;
    v = w[(((size_t)n * Cin + cin) * KH + kh) * KW + kw];
  }
  out[idx] = f32_to_bf16u(v);
}

// ======================================================================
// Fast path: NHWC implicit-GEMM conv, Cin % 32 == 0.
// Block tile 128(M) x 64(N), 8 waves, each wave: 2 A-frag x 2 B-frag ->
// 4 v_wmma_f32_16x16x32_bf16 per K-step. K-step staging = one bounds
// check + 2x16B vector load + 2x16B LDS store per thread (no divisions).
// ======================================================================
template<int KH, int KW>
__global__ __launch_bounds__(256)
void conv_wmma_nhwc(const u16* __restrict__ in,     // bf16 NHWC (N,H,W,Cin)
                    const u16* __restrict__ wfrag,
                    const float* __restrict__ bias,
                    float* __restrict__ out_f32,    // nullable, NHWC (N,H,W,Ctot)+c_off
                    u16*   __restrict__ out_bf,     // nullable, same indexing
                    int Nimg, int Cin, int H, int W,
                    int Cout, int PH, int PW,
                    int Ctot, int c_off, int relu, int KB, int NB) {
  __shared__ __attribute__((aligned(16))) u16 lds[128 * 40];
  const int HW   = H * W;
  const int Mtot = Nimg * HW;
  const int m0   = blockIdx.x * 128;
  const int n0   = blockIdx.y * 64;
  const int tid  = threadIdx.x;
  const int lane = tid & 31;
  const int wid  = tid >> 5;
  const int wm   = (wid & 3) * 32;
  const int wn   = (wid >> 2) * 32;
  const int kbase = (lane < 16) ? 0 : 16;
  const int mrow0 = wm + (lane & 15);
  const int mrow1 = mrow0 + 16;
  const int nbA = (n0 + wn) >> 4;
  const int nbAc = (nbA < NB) ? nbA : 0;
  const int nbBc = (nbA + 1 < NB) ? nbA + 1 : 0;

  // staging: thread t covers (m = t&127, 16 contiguous cin at half = t>>7)
  const int sm   = tid & 127;
  const int half = tid >> 7;           // 0 or 1 -> cin sub-offset 0/16
  int gm = m0 + sm;
  bool mvalid = gm < Mtot;
  int nimg = 0, oh = 0, ow = 0;
  if (mvalid) { nimg = gm / HW; int rem = gm - nimg * HW; oh = rem / W; ow = rem - oh * W; }

  v8f acc00 = {0.f,0.f,0.f,0.f,0.f,0.f,0.f,0.f};
  v8f acc01 = acc00, acc10 = acc00, acc11 = acc00;

  // incremental k decomposition: k = (kh*KW + kw)*Cin + cin0, division-free
  int kh = 0, kw = 0, cin0 = 0;
  for (int kb = 0; kb < KB; ++kb) {
    if (kb + 1 < KB) {
      __builtin_prefetch(wfrag + ((size_t)(kb + 1) * NB + nbAc) * 512 + lane * 16, 0, 1);
      __builtin_prefetch(wfrag + ((size_t)(kb + 1) * NB + nbBc) * 512 + lane * 16, 0, 1);
    }

    // ---- stage A tile: 128 m x 32 cin, vectorized ----
    int ih = oh + kh - PH;
    int iw = ow + kw - PW;
    v8u lo = {0,0,0,0,0,0,0,0}, hi = lo;
    if (mvalid && (unsigned)ih < (unsigned)H && (unsigned)iw < (unsigned)W) {
      const v8u* p = (const v8u*)(in + ((((size_t)nimg * H + ih) * W + iw) * Cin + cin0 + half * 16));
      lo = p[0]; hi = p[1];
    }
    *(v8u*)&lds[sm * 40 + half * 16]     = lo;
    *(v8u*)&lds[sm * 40 + half * 16 + 8] = hi;
    __syncthreads();

    // ---- B fragments: contiguous 32B per lane per subtile ----
    const v8u* bpA = (const v8u*)(wfrag + (((size_t)kb * NB + nbAc) * 512 + lane * 16));
    const v8u* bpB = (const v8u*)(wfrag + (((size_t)kb * NB + nbBc) * 512 + lane * 16));
    v8u bA_lo = bpA[0], bA_hi = bpA[1];
    v8u bB_lo = bpB[0], bB_hi = bpB[1];
    // ---- A fragments from LDS ----
    v8u a0_lo = *(const v8u*)&lds[mrow0 * 40 + kbase];
    v8u a0_hi = *(const v8u*)&lds[mrow0 * 40 + kbase + 8];
    v8u a1_lo = *(const v8u*)&lds[mrow1 * 40 + kbase];
    v8u a1_hi = *(const v8u*)&lds[mrow1 * 40 + kbase + 8];

    v16u a0u, a1u, bAu, bBu;
    #pragma unroll
    for (int j = 0; j < 8; ++j) {
      a0u[j] = a0_lo[j]; a0u[8 + j] = a0_hi[j];
      a1u[j] = a1_lo[j]; a1u[8 + j] = a1_hi[j];
      bAu[j] = bA_lo[j]; bAu[8 + j] = bA_hi[j];
      bBu[j] = bB_lo[j]; bBu[8 + j] = bB_hi[j];
    }
    v16bf a0 = __builtin_bit_cast(v16bf, a0u);
    v16bf a1 = __builtin_bit_cast(v16bf, a1u);
    v16bf bA = __builtin_bit_cast(v16bf, bAu);
    v16bf bB = __builtin_bit_cast(v16bf, bBu);
    acc00 = __builtin_amdgcn_wmma_f32_16x16x32_bf16(false, a0, false, bA, (short)0, acc00, false, false);
    acc01 = __builtin_amdgcn_wmma_f32_16x16x32_bf16(false, a0, false, bB, (short)0, acc01, false, false);
    acc10 = __builtin_amdgcn_wmma_f32_16x16x32_bf16(false, a1, false, bA, (short)0, acc10, false, false);
    acc11 = __builtin_amdgcn_wmma_f32_16x16x32_bf16(false, a1, false, bB, (short)0, acc11, false, false);
    __syncthreads();

    // advance (kh, kw, cin0) by 32 channels
    cin0 += 32;
    if (cin0 >= Cin) {
      cin0 = 0;
      if (++kw == KW) { kw = 0; ++kh; }
    }
  }

  // ---- epilogue: bias + relu, coalesced NHWC stores ----
  #pragma unroll
  for (int mi = 0; mi < 2; ++mi) {
    #pragma unroll
    for (int nj = 0; nj < 2; ++nj) {
      const v8f& acc = mi == 0 ? (nj == 0 ? acc00 : acc01) : (nj == 0 ? acc10 : acc11);
      int nidx = n0 + wn + nj * 16 + (lane & 15);
      bool nok = (nidx < Cout);
      float bv = nok ? bias[nidx] : 0.f;
      #pragma unroll
      for (int r = 0; r < 8; ++r) {
        int m = m0 + wm + mi * 16 + r + ((lane >> 4) << 3);
        if (nok && m < Mtot) {
          float v = acc[r] + bv;
          if (relu) v = fmaxf(v, 0.f);
          size_t oidx = (size_t)m * Ctot + c_off + nidx;
          if (out_f32) out_f32[oidx] = v;
          if (out_bf)  out_bf[oidx]  = f32_to_bf16u(v);
        }
      }
    }
  }
}

// ======================================================================
// Gather path (layer 1 only: Cin=3, K=27, single K-step).
// ======================================================================
template<int KH, int KW, int CIN>
__global__ __launch_bounds__(256)
void conv_wmma_gather(const u16* __restrict__ in,    // bf16 NHWC
                      const u16* __restrict__ wfrag,
                      const float* __restrict__ bias,
                      u16* __restrict__ out_bf,      // NHWC (N,H,W,Cout)
                      int Nimg, int H, int W,
                      int Cout, int PH, int PW, int relu, int KB, int NB) {
  constexpr int K = CIN * KH * KW;
  __shared__ __attribute__((aligned(16))) u16 lds[128 * 40];
  const int HW   = H * W;
  const int Mtot = Nimg * HW;
  const int m0   = blockIdx.x * 128;
  const int n0   = blockIdx.y * 64;
  const int tid  = threadIdx.x;
  const int lane = tid & 31;
  const int wid  = tid >> 5;
  const int wm   = (wid & 3) * 32;
  const int wn   = (wid >> 2) * 32;
  const int kbase = (lane < 16) ? 0 : 16;
  const int mrow0 = wm + (lane & 15);
  const int mrow1 = mrow0 + 16;
  const int nbA = (n0 + wn) >> 4;
  const int nbAc = (nbA < NB) ? nbA : 0;
  const int nbBc = (nbA + 1 < NB) ? nbA + 1 : 0;

  const int sm  = tid & 127;
  const int sk0 = tid >> 7;
  int gm = m0 + sm;
  bool mvalid = gm < Mtot;
  int nimg = 0, oh = 0, ow = 0;
  if (mvalid) { nimg = gm / HW; int rem = gm - nimg * HW; oh = rem / W; ow = rem - oh * W; }

  v8f acc00 = {0.f,0.f,0.f,0.f,0.f,0.f,0.f,0.f};
  v8f acc01 = acc00, acc10 = acc00, acc11 = acc00;

  for (int kb = 0; kb < KB; ++kb) {
    #pragma unroll
    for (int i = 0; i < 16; ++i) {
      int kl = sk0 + i * 2;
      int k  = kb * 32 + kl;
      u16 hv = 0;
      if (mvalid && k < K) {
        int khkw = k / CIN; int cin = k - khkw * CIN;   // constant divisors
        int kh = khkw / KW; int kw = khkw - kh * KW;
        int ih = oh + kh - PH, iw = ow + kw - PW;
        if ((unsigned)ih < (unsigned)H && (unsigned)iw < (unsigned)W)
          hv = in[(((size_t)nimg * H + ih) * W + iw) * CIN + cin];
      }
      lds[sm * 40 + kl] = hv;
    }
    __syncthreads();

    const v8u* bpA = (const v8u*)(wfrag + (((size_t)kb * NB + nbAc) * 512 + lane * 16));
    const v8u* bpB = (const v8u*)(wfrag + (((size_t)kb * NB + nbBc) * 512 + lane * 16));
    v8u bA_lo = bpA[0], bA_hi = bpA[1];
    v8u bB_lo = bpB[0], bB_hi = bpB[1];
    v8u a0_lo = *(const v8u*)&lds[mrow0 * 40 + kbase];
    v8u a0_hi = *(const v8u*)&lds[mrow0 * 40 + kbase + 8];
    v8u a1_lo = *(const v8u*)&lds[mrow1 * 40 + kbase];
    v8u a1_hi = *(const v8u*)&lds[mrow1 * 40 + kbase + 8];

    v16u a0u, a1u, bAu, bBu;
    #pragma unroll
    for (int j = 0; j < 8; ++j) {
      a0u[j] = a0_lo[j]; a0u[8 + j] = a0_hi[j];
      a1u[j] = a1_lo[j]; a1u[8 + j] = a1_hi[j];
      bAu[j] = bA_lo[j]; bAu[8 + j] = bA_hi[j];
      bBu[j] = bB_lo[j]; bBu[8 + j] = bB_hi[j];
    }
    v16bf a0 = __builtin_bit_cast(v16bf, a0u);
    v16bf a1 = __builtin_bit_cast(v16bf, a1u);
    v16bf bA = __builtin_bit_cast(v16bf, bAu);
    v16bf bB = __builtin_bit_cast(v16bf, bBu);
    acc00 = __builtin_amdgcn_wmma_f32_16x16x32_bf16(false, a0, false, bA, (short)0, acc00, false, false);
    acc01 = __builtin_amdgcn_wmma_f32_16x16x32_bf16(false, a0, false, bB, (short)0, acc01, false, false);
    acc10 = __builtin_amdgcn_wmma_f32_16x16x32_bf16(false, a1, false, bA, (short)0, acc10, false, false);
    acc11 = __builtin_amdgcn_wmma_f32_16x16x32_bf16(false, a1, false, bB, (short)0, acc11, false, false);
    __syncthreads();
  }

  #pragma unroll
  for (int mi = 0; mi < 2; ++mi) {
    #pragma unroll
    for (int nj = 0; nj < 2; ++nj) {
      const v8f& acc = mi == 0 ? (nj == 0 ? acc00 : acc01) : (nj == 0 ? acc10 : acc11);
      int nidx = n0 + wn + nj * 16 + (lane & 15);
      bool nok = (nidx < Cout);
      float bv = nok ? bias[nidx] : 0.f;
      #pragma unroll
      for (int r = 0; r < 8; ++r) {
        int m = m0 + wm + mi * 16 + r + ((lane >> 4) << 3);
        if (nok && m < Mtot) {
          float v = acc[r] + bv;
          if (relu) v = fmaxf(v, 0.f);
          out_bf[(size_t)m * Cout + nidx] = f32_to_bf16u(v);
        }
      }
    }
  }
}

// ---------------- 2x2 maxpool (bf16, NHWC) ----------------
__global__ void maxpool2(const u16* __restrict__ in, u16* __restrict__ out,
                         int N, int C, int Ho, int Wo) {
  int idx = blockIdx.x * blockDim.x + threadIdx.x;
  int total = N * C * Ho * Wo;
  if (idx >= total) return;
  int c = idx % C; int t = idx / C;
  int w = t % Wo; t /= Wo;
  int h = t % Ho; int n = t / Ho;
  int Wi = Wo * 2;
  size_t base = (((size_t)n * (Ho * 2) + h * 2) * Wi + w * 2) * C + c;
  size_t rs = (size_t)Wi * C;
  float a = bf16u_to_f32(in[base]),      b = bf16u_to_f32(in[base + C]);
  float d = bf16u_to_f32(in[base + rs]), e = bf16u_to_f32(in[base + rs + C]);
  out[idx] = f32_to_bf16u(fmaxf(fmaxf(a, b), fmaxf(d, e)));
}

// ---------------- directional relu-add scan (in place, fp32 NHWC) ----------------
__global__ void scan_dir(float* __restrict__ buf, int N, int Ctot, int c_off, int C,
                         int H, int W, int axis, int rev) {
  int lines = (axis == 2) ? N * C * W : N * C * H;
  int tid = blockIdx.x * blockDim.x + threadIdx.x;
  if (tid >= lines) return;
  size_t base; int len; long stride;
  if (axis == 2) {          // scan along H
    int c = tid % C; int t = tid / C; int w = t % W; int n = t / W;
    base = ((size_t)n * H * W + w) * Ctot + c_off + c;
    len = H; stride = (long)W * Ctot;
  } else {                  // scan along W
    int c = tid % C; int t = tid / C; int h = t % H; int n = t / H;
    base = (((size_t)n * H + h) * W) * (size_t)Ctot + c_off + c;
    len = W; stride = Ctot;
  }
  long idx  = (long)base + (rev ? (long)(len - 1) * stride : 0);
  long step = rev ? -stride : stride;
  float prev = buf[idx];
  for (int i = 1; i < len; ++i) {
    idx += step;
    float v = fmaxf(buf[idx] + prev, 0.f);
    buf[idx] = v;
    prev = v;
  }
}

// ---------------- bilinear resize: NHWC fp32 in -> NCHW fp32 out ----------------
__global__ void resize_bilinear(const float* __restrict__ in, float* __restrict__ out,
                                int N, int C, int Hi, int Wi, int Ho, int Wo) {
  int idx = blockIdx.x * blockDim.x + threadIdx.x;
  int total = N * C * Ho * Wo;
  if (idx >= total) return;
  int wo = idx % Wo; int t = idx / Wo;
  int ho = t % Ho; t /= Ho;
  int c = t % C; int n = t / C;
  float sh = (float)Hi / (float)Ho, sw = (float)Wi / (float)Wo;
  float fh = (ho + 0.5f) * sh - 0.5f;
  float fw = (wo + 0.5f) * sw - 0.5f;
  int h0 = (int)floorf(fh); float th = fh - (float)h0;
  int w0 = (int)floorf(fw); float tw = fw - (float)w0;
  int h1 = h0 + 1, w1 = w0 + 1;
  h0 = h0 < 0 ? 0 : (h0 > Hi - 1 ? Hi - 1 : h0);
  h1 = h1 < 0 ? 0 : (h1 > Hi - 1 ? Hi - 1 : h1);
  w0 = w0 < 0 ? 0 : (w0 > Wi - 1 ? Wi - 1 : w0);
  w1 = w1 < 0 ? 0 : (w1 > Wi - 1 ? Wi - 1 : w1);
  const float* p = in + (size_t)n * Hi * Wi * C + c;
  float v00 = p[((size_t)h0 * Wi + w0) * C], v01 = p[((size_t)h0 * Wi + w1) * C];
  float v10 = p[((size_t)h1 * Wi + w0) * C], v11 = p[((size_t)h1 * Wi + w1) * C];
  out[idx] = v00 * (1.f - th) * (1.f - tw) + v01 * (1.f - th) * tw +
             v10 * th * (1.f - tw) + v11 * th * tw;
}

// ---------------- host orchestration ----------------
extern "C" void kernel_launch(void* const* d_in, const int* in_sizes, int n_in,
                              void* d_out, int out_size, void* d_ws, size_t ws_size,
                              hipStream_t stream) {
  (void)in_sizes; (void)n_in; (void)out_size; (void)ws_size;
  const float* x = (const float*)d_in[0];
  auto WP = [&](int i) { return (const float*)d_in[i]; };

  char* ws = (char*)d_ws;
  size_t off = 0;
  auto alloc = [&](size_t bytes) -> void* {
    void* p = ws + off;
    off = (off + bytes + 255) & ~(size_t)255;
    return p;
  };
  const size_t ACT = (size_t)2 * 64 * 288 * 800;           // max activation elems
  u16*   actA   = (u16*)alloc(ACT * 2);
  u16*   actB   = (u16*)alloc(ACT * 2);
  u16*   wfrag  = (u16*)alloc((size_t)144 * 64 * 512 * 2); // max (cls1) fragments
  float* concat = (float*)alloc((size_t)2 * 512 * 36 * 100 * 4);
  float* cls2o  = (float*)alloc((size_t)2 * 4 * 36 * 100 * 4);

  auto repack = [&](const float* w, int Cin, int Cout, int KH, int KW, int& KB, int& NB) {
    int K = Cin * KH * KW;
    KB = (K + 31) / 32;
    NB = (Cout + 15) / 16;
    int rp = KB * NB * 512;
    repack_w<<<(rp + 255) / 256, 256, 0, stream>>>(w, wfrag, Cout, Cin, KH, KW, KB, NB);
  };

  auto conv = [&]<int KH, int KW>(const u16* in, const float* w, const float* b,
                                  int Cin, int Cout, int H, int Wd, int PH, int PW,
                                  float* of32, u16* obf, int Ctot, int c_off, int relu) {
    int KB, NB;
    repack(w, Cin, Cout, KH, KW, KB, NB);
    int Mtot = 2 * H * Wd;
    dim3 grid((Mtot + 127) / 128, (Cout + 63) / 64);
    conv_wmma_nhwc<KH, KW><<<grid, 256, 0, stream>>>(in, wfrag, b, of32, obf,
                                                     2, Cin, H, Wd, Cout, PH, PW,
                                                     Ctot, c_off, relu, KB, NB);
  };

  // x: NCHW fp32 -> NHWC bf16
  {
    int n = 2 * 288 * 800 * 3;
    cvt_nchw_nhwc<<<(n + 255) / 256, 256, 0, stream>>>(x, actA, 2, 3, 288, 800);
  }

  u16* cur = actA; u16* nxt = actB;
  int H = 288, Wd = 800;

  // layer 1: Cin=3 gather path
  {
    int KB, NB;
    repack(WP(1), 3, 64, 3, 3, KB, NB);
    int Mtot = 2 * H * Wd;
    dim3 grid((Mtot + 127) / 128, 1);
    conv_wmma_gather<3, 3, 3><<<grid, 256, 0, stream>>>(cur, wfrag, WP(2), nxt,
                                                        2, H, Wd, 64, 1, 1, 1, KB, NB);
    u16* t = cur; cur = nxt; nxt = t;
  }

  const int fd[13][2] = {{3,64},{64,64},{64,128},{128,128},{128,256},{256,256},{256,256},
                         {256,512},{512,512},{512,512},{512,512},{512,512},{512,512}};
  const bool pool_after[13] = {false,true,false,true,false,false,true,
                               false,false,false,false,false,false};
  for (int i = 1; i < 13; ++i) {
    conv.template operator()<3, 3>(cur, WP(1 + 2 * i), WP(2 + 2 * i),
                                   fd[i][0], fd[i][1], H, Wd, 1, 1,
                                   nullptr, nxt, fd[i][1], 0, 1);
    u16* t = cur; cur = nxt; nxt = t;
    if (pool_after[i]) {
      H /= 2; Wd /= 2;
      int total = 2 * fd[i][1] * H * Wd;
      maxpool2<<<(total + 255) / 256, 256, 0, stream>>>(cur, nxt, 2, fd[i][1], H, Wd);
      t = cur; cur = nxt; nxt = t;
    }
  }
  // backbone output: cur = bf16 NHWC (2,36,100,512)

  // SCNN directional convs -> concat slices (fp32 NHWC, Ctot=512)
  conv.template operator()<1, 9>(cur, WP(27), WP(28), 512, 128, 36, 100, 0, 4,
                                 concat, nullptr, 512,   0, 0);
  conv.template operator()<1, 9>(cur, WP(29), WP(30), 512, 128, 36, 100, 0, 4,
                                 concat, nullptr, 512, 128, 0);
  conv.template operator()<9, 1>(cur, WP(31), WP(32), 512, 128, 36, 100, 4, 0,
                                 concat, nullptr, 512, 256, 0);
  conv.template operator()<9, 1>(cur, WP(33), WP(34), 512, 128, 36, 100, 4, 0,
                                 concat, nullptr, 512, 384, 0);

  // directional scans (d: H fwd, u: H rev, r: W fwd, l: W rev)
  {
    int lh = 2 * 128 * 100, lw = 2 * 128 * 36;
    scan_dir<<<(lh + 255) / 256, 256, 0, stream>>>(concat, 2, 512,   0, 128, 36, 100, 2, 0);
    scan_dir<<<(lh + 255) / 256, 256, 0, stream>>>(concat, 2, 512, 128, 128, 36, 100, 2, 1);
    scan_dir<<<(lw + 255) / 256, 256, 0, stream>>>(concat, 2, 512, 256, 128, 36, 100, 3, 0);
    scan_dir<<<(lw + 255) / 256, 256, 0, stream>>>(concat, 2, 512, 384, 128, 36, 100, 3, 1);
  }

  // concat fp32 -> bf16 (same NHWC layout)
  {
    int n = 2 * 36 * 100 * 512;
    cvt_f32_bf16<<<(n + 255) / 256, 256, 0, stream>>>(concat, cur, n);
  }

  // classifier
  conv.template operator()<3, 3>(cur, WP(35), WP(36), 512, 1024, 36, 100, 1, 1,
                                 nullptr, nxt, 1024, 0, 1);
  conv.template operator()<1, 1>(nxt, WP(37), WP(38), 1024, 4, 36, 100, 0, 0,
                                 cls2o, nullptr, 4, 0, 0);

  // bilinear upsample NHWC(2,36,100,4) -> NCHW (2,4,288,800)
  {
    int total = 2 * 4 * 288 * 800;
    resize_bilinear<<<(total + 255) / 256, 256, 0, stream>>>(cls2o, (float*)d_out,
                                                             2, 4, 36, 100, 288, 800);
  }
}
